// SimpleTGN_38577396252850
// MI455X (gfx1250) — matrix operand order repaired and checked by reading, not compiled, not run on hardware
//
#include <hip/hip_runtime.h>
#include <math.h>

typedef __attribute__((ext_vector_type(16))) _Float16 v16h;
typedef __attribute__((ext_vector_type(8)))  _Float16 v8h;
typedef __attribute__((ext_vector_type(8)))  float    v8f;

#define IN_DIM 128
#define HID    256
#define NGR    128

// ---------------- utility kernels ----------------

__global__ void fill_f32(float* __restrict__ p, float v, long long n) {
    long long i = (long long)blockIdx.x * blockDim.x + threadIdx.x;
    if (i < n) p[i] = v;
}

__global__ void cvt_f32_to_f16(const float* __restrict__ in, _Float16* __restrict__ out,
                               long long n) {
    long long i = (long long)blockIdx.x * blockDim.x + threadIdx.x;
    if (i < n) out[i] = (_Float16)in[i];
}

// deg accumulate: dinv buffer pre-filled with 1.0 (self loop), add 1 per in-edge
__global__ void deg_accum(const int* __restrict__ dst, float* __restrict__ deg, int E) {
    int e = blockIdx.x * blockDim.x + threadIdx.x;
    if (e < E) atomicAdd(&deg[dst[e]], 1.0f);
}

__global__ void deg_to_dinv(float* __restrict__ d, int n) {
    int i = blockIdx.x * blockDim.x + threadIdx.x;
    if (i < n) d[i] = rsqrtf(d[i]);   // deg >= 1 always (self loop)
}

// ---------------- WMMA GEMM: C[f32, MxN] = A[f16, MxK] @ B[f16, KxN] ----------------
// Each wave computes a 16x64 strip: one A fragment per K-step reused across 4 B tiles
// (4x less L2 traffic on the big activation stream). 4 waves/block -> 16x256 per block.
// M guarded; K mult of 32; N mult of 64.
__global__ void wmma_gemm_f16(const _Float16* __restrict__ A,
                              const _Float16* __restrict__ B,
                              float* __restrict__ C,
                              int M, int N, int K) {
    const int lane = threadIdx.x & 31;
    const int wave = threadIdx.x >> 5;
    const int m0   = blockIdx.x * 16;
    const int n0   = (blockIdx.y * 4 + wave) * 64;
    if (n0 >= N) return;                       // wave-uniform: EXEC stays all-1s

    const int grp  = lane >> 4;                // 0: K chunks {0..7,16..23}; 1: {8..15,24..31}
    int mrow = m0 + (lane & 15);
    if (mrow >= M) mrow = M - 1;               // clamp loads; stores guarded below

    v8f acc0 = {}, acc1 = {}, acc2 = {}, acc3 = {};
    for (int k0 = 0; k0 < K; k0 += 32) {
        // A fragment: row mrow, halves [k0+8g, k0+8g+8) and [k0+16+8g, k0+16+8g+8)
        const _Float16* ap = A + (size_t)mrow * K + k0 + 8 * grp;
        v8h alo = *(const v8h*)(ap);
        v8h ahi = *(const v8h*)(ap + 16);
        v16h afrag = __builtin_shufflevector(alo, ahi,
            0,1,2,3,4,5,6,7,8,9,10,11,12,13,14,15);
        // B fragments: lane holds column n0+16j+(lane&15), K rows k0+8g+{0..7} and k0+16+8g+{0..7}
        const _Float16* bp = B + (size_t)(k0 + 8 * grp) * N + n0 + (lane & 15);
        #pragma unroll
        for (int j = 0; j < 4; ++j) {
            v16h bfrag;
            #pragma unroll
            for (int s = 0; s < 8; ++s) {
                bfrag[s]     = bp[(size_t)s * N + 16 * j];
                bfrag[8 + s] = bp[(size_t)(16 + s) * N + 16 * j];
            }
            v8f* accp = (j == 0) ? &acc0 : (j == 1) ? &acc1 : (j == 2) ? &acc2 : &acc3;
            *accp = __builtin_amdgcn_wmma_f32_16x16x32_f16(
                /*neg_a=*/false, afrag, /*neg_b=*/false, bfrag,
                /*c_mod=*/(short)0, *accp, /*reuse_a=*/false, /*reuse_b=*/false);
        }
    }
    // D layout: VGPR r -> row m0 + r + 8*grp, col n_tile_base + (lane&15)
    #pragma unroll
    for (int j = 0; j < 4; ++j) {
        const v8f* accp = (j == 0) ? &acc0 : (j == 1) ? &acc1 : (j == 2) ? &acc2 : &acc3;
        const int ncol = n0 + 16 * j + (lane & 15);
        #pragma unroll
        for (int r = 0; r < 8; ++r) {
            int mr = m0 + r + 8 * grp;
            if (mr < M) C[(size_t)mr * N + ncol] = (*accp)[r];
        }
    }
}

// ---------------- edge scatter (the L2-bound hot loop) ----------------
// one thread per (edge, 4-feature chunk): float4 gather + 4 atomic adds
__global__ void scatter_edges(const int* __restrict__ src, const int* __restrict__ dst,
                              const float* __restrict__ dinv,
                              const float* __restrict__ h, float* __restrict__ acc,
                              int E, int F4) {
    long long idx = (long long)blockIdx.x * blockDim.x + threadIdx.x;
    long long total = (long long)E * F4;
    if (idx >= total) return;
    int e = (int)(idx / F4);
    int c = (int)(idx - (long long)e * F4) * 4;
    int s = src[e], d = dst[e];
    float norm = dinv[s] * dinv[d];
    const float4 v = *(const float4*)(h + (size_t)s * (F4 * 4) + c);
    float* o = acc + (size_t)d * (F4 * 4) + c;
    atomicAdd(o + 0, v.x * norm);
    atomicAdd(o + 1, v.y * norm);
    atomicAdd(o + 2, v.z * norm);
    atomicAdd(o + 3, v.w * norm);
}

// finalize: add self-loop term + bias, relu; write f32 (in place ok) and f16 copy
__global__ void finalize_layer(const float* __restrict__ acc, const float* __restrict__ hw,
                               const float* __restrict__ dinv, const float* __restrict__ bias,
                               float* __restrict__ out32, _Float16* __restrict__ out16,
                               int Nn, int F) {
    long long i = (long long)blockIdx.x * blockDim.x + threadIdx.x;
    if (i >= (long long)Nn * F) return;
    int n = (int)(i / F);
    int f = (int)(i - (long long)n * F);
    float di = dinv[n];
    float v = acc[i] + hw[i] * di * di + bias[f];
    v = fmaxf(v, 0.0f);
    out32[i] = v;
    out16[i] = (_Float16)v;
}

// ---------------- pooling ----------------
__global__ void pool_accum(const float* __restrict__ h2, const int* __restrict__ batch,
                           float* __restrict__ gsum, int Nn) {
    long long idx = (long long)blockIdx.x * blockDim.x + threadIdx.x;
    const int F4 = HID / 4;
    if (idx >= (long long)Nn * F4) return;
    int n = (int)(idx / F4);
    int c = (int)(idx - (long long)n * F4) * 4;
    int b = batch[n];
    const float4 v = *(const float4*)(h2 + (size_t)n * HID + c);
    float* o = gsum + (size_t)b * HID + c;
    atomicAdd(o + 0, v.x);
    atomicAdd(o + 1, v.y);
    atomicAdd(o + 2, v.z);
    atomicAdd(o + 3, v.w);
}

__global__ void pool_count(const int* __restrict__ batch, float* __restrict__ cnt, int Nn) {
    int n = blockIdx.x * blockDim.x + threadIdx.x;
    if (n < Nn) atomicAdd(&cnt[batch[n]], 1.0f);
}

__global__ void pool_div(const float* __restrict__ gsum, const float* __restrict__ cnt,
                         _Float16* __restrict__ g16, int total) {
    int i = blockIdx.x * blockDim.x + threadIdx.x;
    if (i >= total) return;
    int g = i / HID;
    float c = fmaxf(cnt[g], 1.0f);
    g16[i] = (_Float16)(gsum[i] / c);
}

// ---------------- classifier head: sigmoid(relu(z+b3) @ W4 + b4) ----------------
__global__ void head_kernel(const float* __restrict__ z, const float* __restrict__ b3,
                            const float* __restrict__ W4, const float* __restrict__ b4,
                            float* __restrict__ out) {
    int g = blockIdx.x * blockDim.x + threadIdx.x;
    if (g >= NGR) return;
    float s = b4[0];
    const int H2 = HID / 2;
    for (int j = 0; j < H2; ++j)
        s += fmaxf(z[(size_t)g * H2 + j] + b3[j], 0.0f) * W4[j];
    out[g] = 1.0f / (1.0f + expf(-s));
}

// ---------------- host driver ----------------
static inline unsigned blks(long long n, int b) { return (unsigned)((n + b - 1) / b); }

extern "C" void kernel_launch(void* const* d_in, const int* in_sizes, int n_in,
                              void* d_out, int out_size, void* d_ws, size_t ws_size,
                              hipStream_t stream) {
    const float* x     = (const float*)d_in[0];
    const int*   ei    = (const int*)d_in[1];
    const int*   batch = (const int*)d_in[2];
    const float* W1 = (const float*)d_in[3];
    const float* b1 = (const float*)d_in[4];
    const float* W2 = (const float*)d_in[5];
    const float* b2 = (const float*)d_in[6];
    const float* W3 = (const float*)d_in[7];
    const float* b3 = (const float*)d_in[8];
    const float* W4 = (const float*)d_in[9];
    const float* b4 = (const float*)d_in[10];
    float* out = (float*)d_out;

    const int Nn = in_sizes[0] / IN_DIM;   // 50000
    const int E  = in_sizes[1] / 2;        // 800000
    const int* src = ei;
    const int* dst = ei + E;

    // carve workspace
    char* p = (char*)d_ws;
    auto alloc = [&](size_t bytes) -> char* {
        char* r = p; p += (bytes + 255) & ~(size_t)255; return r;
    };
    _Float16* Ah   = (_Float16*)alloc((size_t)Nn * HID * sizeof(_Float16)); // f16 activations
    float*    Bf   = (float*)   alloc((size_t)Nn * HID * sizeof(float));    // GEMM out (hW)
    float*    Cf   = (float*)   alloc((size_t)Nn * HID * sizeof(float));    // scatter acc
    float*    dinv = (float*)   alloc((size_t)Nn * sizeof(float));
    _Float16* W1h  = (_Float16*)alloc((size_t)IN_DIM * HID * sizeof(_Float16));
    _Float16* W2h  = (_Float16*)alloc((size_t)HID * HID * sizeof(_Float16));
    _Float16* W3h  = (_Float16*)alloc((size_t)HID * (HID / 2) * sizeof(_Float16));
    float*    gsum = (float*)   alloc((size_t)NGR * HID * sizeof(float));
    float*    gcnt = (float*)   alloc((size_t)NGR * sizeof(float));
    _Float16* g16  = (_Float16*)alloc((size_t)NGR * HID * sizeof(_Float16));
    float*    z    = (float*)   alloc((size_t)NGR * (HID / 2) * sizeof(float));
    (void)ws_size; (void)n_in; (void)out_size;

    const long long NF   = (long long)Nn * HID;
    const long long EF4  = (long long)E * (HID / 4);

    // degrees -> dinv (self loop included via init = 1.0)
    fill_f32<<<blks(Nn, 256), 256, 0, stream>>>(dinv, 1.0f, Nn);
    deg_accum<<<blks(E, 256), 256, 0, stream>>>(dst, dinv, E);
    deg_to_dinv<<<blks(Nn, 256), 256, 0, stream>>>(dinv, Nn);

    // f16 conversions
    cvt_f32_to_f16<<<blks((long long)Nn * IN_DIM, 256), 256, 0, stream>>>(x, Ah, (long long)Nn * IN_DIM);
    cvt_f32_to_f16<<<blks(IN_DIM * HID, 256), 256, 0, stream>>>(W1, W1h, IN_DIM * HID);
    cvt_f32_to_f16<<<blks(HID * HID, 256), 256, 0, stream>>>(W2, W2h, HID * HID);
    cvt_f32_to_f16<<<blks(HID * (HID / 2), 256), 256, 0, stream>>>(W3, W3h, HID * (HID / 2));

    // ---- layer 1: hW = x @ W1 ; scatter ; h1 = relu(agg + self + b1) ----
    {
        dim3 grid(blks(Nn, 16), blks(HID, 256));       // 16x64 strip per wave
        wmma_gemm_f16<<<grid, 128, 0, stream>>>(Ah, W1h, Bf, Nn, HID, IN_DIM);
    }
    fill_f32<<<blks(NF, 256), 256, 0, stream>>>(Cf, 0.0f, NF);
    scatter_edges<<<blks(EF4, 256), 256, 0, stream>>>(src, dst, dinv, Bf, Cf, E, HID / 4);
    finalize_layer<<<blks(NF, 256), 256, 0, stream>>>(Cf, Bf, dinv, b1, Cf, Ah, Nn, HID);

    // ---- layer 2: hW = h1 @ W2 ; scatter ; h2 = relu(...) (f32 into Bf for pooling) ----
    {
        dim3 grid(blks(Nn, 16), blks(HID, 256));
        wmma_gemm_f16<<<grid, 128, 0, stream>>>(Ah, W2h, Bf, Nn, HID, HID);
    }
    fill_f32<<<blks(NF, 256), 256, 0, stream>>>(Cf, 0.0f, NF);
    scatter_edges<<<blks(EF4, 256), 256, 0, stream>>>(src, dst, dinv, Bf, Cf, E, HID / 4);
    finalize_layer<<<blks(NF, 256), 256, 0, stream>>>(Cf, Bf, dinv, b2, Bf, Ah, Nn, HID);

    // ---- global mean pool ----
    fill_f32<<<blks(NGR * HID, 256), 256, 0, stream>>>(gsum, 0.0f, NGR * HID);
    fill_f32<<<1, 256, 0, stream>>>(gcnt, 0.0f, NGR);
    pool_accum<<<blks((long long)Nn * (HID / 4), 256), 256, 0, stream>>>(Bf, batch, gsum, Nn);
    pool_count<<<blks(Nn, 256), 256, 0, stream>>>(batch, gcnt, Nn);
    pool_div<<<blks(NGR * HID, 256), 256, 0, stream>>>(gsum, gcnt, g16, NGR * HID);

    // ---- classifier: z = g @ W3 (WMMA), then sigmoid(relu(z+b3)@W4 + b4) ----
    {
        dim3 grid(blks(NGR, 16), blks(HID / 2, 256));  // N=128 -> waves 2,3 idle (uniform)
        wmma_gemm_f16<<<grid, 128, 0, stream>>>(g16, W3h, z, NGR, HID / 2, HID);
    }
    head_kernel<<<1, 128, 0, stream>>>(z, b3, W4, b4, out);
}
